// LayerNormGRUCell_25838523253290
// MI455X (gfx1250) — matrix operand order repaired and checked by reading, not compiled, not run on hardware
//
#include <hip/hip_runtime.h>

// ---------------------------------------------------------------------------
// LayerNorm GRU cell, B=8192, IN=H=1024, fp32 in/out.
// bf16 WMMA GEMMs (v_wmma_f32_16x16x32_bf16) with f32 accumulation.
// GEMM operands staged in LDS via async global->LDS copies (ASYNCcnt),
// double buffered; fragments read from LDS with ds_load_b128.
// ---------------------------------------------------------------------------

#define BB   8192
#define HH   1024
#define KK   1024

#define BKS  64      // K-step per LDS stage
#define LDA  72      // padded LDS row stride (elements) -> 144B, conflict-free

typedef __attribute__((ext_vector_type(16))) __bf16 v16bf;
typedef __attribute__((ext_vector_type(8)))  float  v8f;
typedef __attribute__((ext_vector_type(4)))  int    v4i;

union FragBF16 {
    uint4 q[2];   // 32 bytes
    v16bf v;      // 16 x bf16 per lane
};

#if defined(__gfx1250__) && \
    __has_builtin(__builtin_amdgcn_global_load_async_to_lds_b128) && \
    __has_builtin(__builtin_amdgcn_s_wait_asynccnt)
#define USE_ASYNC_LDS 1
#else
#define USE_ASYNC_LDS 0
#endif

__device__ __forceinline__ void async_cp_b128(const unsigned short* g, unsigned short* l) {
#if USE_ASYNC_LDS
    // GLOBAL_LOAD_ASYNC_TO_LDS_B128: memory -> LDS, tracked with ASYNCcnt
    __builtin_amdgcn_global_load_async_to_lds_b128(
        (__attribute__((address_space(1))) v4i*)g,
        (__attribute__((address_space(3))) v4i*)l,
        0, 0);
#else
    *(uint4*)l = *(const uint4*)g;
#endif
}

__device__ __forceinline__ void wait_async_all() {
#if USE_ASYNC_LDS
    __builtin_amdgcn_s_wait_asynccnt(0);
#endif
}

__device__ __forceinline__ unsigned short f32_to_bf16(float f) {
    unsigned int u = __float_as_uint(f);
    u += 0x7FFFu + ((u >> 16) & 1u);   // round-to-nearest-even
    return (unsigned short)(u >> 16);
}

// ---------------------------------------------------------------------------
// fp32 -> bf16 conversion (vectorized x4)
// ---------------------------------------------------------------------------
__global__ __launch_bounds__(256)
void cvt_f32_bf16(const float* __restrict__ src, unsigned short* __restrict__ dst, int n4) {
    int i = blockIdx.x * blockDim.x + threadIdx.x;
    if (i >= n4) return;
    float4 f = ((const float4*)src)[i];
    unsigned int lo = ((unsigned int)f32_to_bf16(f.y) << 16) | f32_to_bf16(f.x);
    unsigned int hi = ((unsigned int)f32_to_bf16(f.w) << 16) | f32_to_bf16(f.z);
    ((uint2*)dst)[i] = make_uint2(lo, hi);
}

// ---------------------------------------------------------------------------
// GEMM: Y[M,N] = A[M,K] @ W[N,K]^T   (A,W bf16 row-major, Y f32)
// 128x128 block tile, 8 waves (4 row groups x 2 col groups),
// each wave: 32x64 = 2x4 tiles of 16x16.
// K loop: stages of 64 staged in double-buffered LDS via async copies,
// two WMMA substeps (K=32) per stage.
//
// Fragment layouts as used here (wave32; A/B use contiguous k = 16h + s, same in both):
//  A 16x32 bf16 : lane m(0..15): row m, K=0..15; lane m+16: K=16..31
//  B 32x16 bf16 : lane l(0..15): col l, K=0..15; lane l+16: col l, K=16..31
//  C/D 16x16 f32: vgpr i, lanes 0..15 -> (M=i, N=lane); lanes 16..31 -> (M=8+i)
// ---------------------------------------------------------------------------
__global__ __launch_bounds__(256)
void gemm_bf16_wmma(const unsigned short* __restrict__ A,
                    const unsigned short* __restrict__ Wm,
                    float* __restrict__ Y,
                    int Kd, int N) {
    __shared__ unsigned short Al[2][128 * LDA];
    __shared__ unsigned short Bl[2][128 * LDA];

    const int tid    = threadIdx.x;
    const int lane   = tid & 31;
    const int wave   = tid >> 5;
    const int wr     = wave & 3;          // 0..3  (row group of 32)
    const int wc     = wave >> 2;         // 0..1  (col group of 64)
    const int laneLo = lane & 15;
    const int hi     = lane >> 4;

    const int rowBlk = blockIdx.y * 128;
    const int colBlk = blockIdx.x * 128;

    v8f acc[2][4] = {};

    // stage copy: 128 rows x 64 bf16 each for A and B; 8 b128 chunks per row;
    // 1024 chunks per tile -> 4 per thread per tile.
    auto issue_stage = [&](int s, int buf) {
        const int k0 = s * BKS;
        #pragma unroll
        for (int i = 0; i < 4; ++i) {
            int l = tid + i * 256;
            int row = l >> 3, ch = l & 7;
            async_cp_b128(A + (size_t)(rowBlk + row) * Kd + k0 + ch * 8,
                          &Al[buf][row * LDA + ch * 8]);
        }
        #pragma unroll
        for (int i = 0; i < 4; ++i) {
            int l = tid + i * 256;
            int row = l >> 3, ch = l & 7;
            async_cp_b128(Wm + (size_t)(colBlk + row) * Kd + k0 + ch * 8,
                          &Bl[buf][row * LDA + ch * 8]);
        }
    };

    const int nstages = Kd / BKS;
    issue_stage(0, 0);

    for (int s = 0; s < nstages; ++s) {
        const int buf = s & 1;
        wait_async_all();       // this thread's stage-s copies done
        __syncthreads();        // everyone's copies done; stage s-1 compute done
        if (s + 1 < nstages) issue_stage(s + 1, buf ^ 1);

        const unsigned short* Ab = &Al[buf][0];
        const unsigned short* Bb = &Bl[buf][0];
        #pragma unroll
        for (int kk = 0; kk < BKS; kk += 32) {
            FragBF16 a0, a1, b0, b1, b2, b3;
            const unsigned short* ap0 = Ab + (wr * 32 + laneLo) * LDA + kk + hi * 16;
            const unsigned short* ap1 = ap0 + 16 * LDA;
            a0.q[0] = *(const uint4*)(ap0);
            a0.q[1] = *(const uint4*)(ap0 + 8);
            a1.q[0] = *(const uint4*)(ap1);
            a1.q[1] = *(const uint4*)(ap1 + 8);

            const unsigned short* bp0 = Bb + (wc * 64 +  0 + laneLo) * LDA + kk + hi * 16;
            const unsigned short* bp1 = bp0 + 16 * LDA;
            const unsigned short* bp2 = bp0 + 32 * LDA;
            const unsigned short* bp3 = bp0 + 48 * LDA;
            b0.q[0] = *(const uint4*)(bp0); b0.q[1] = *(const uint4*)(bp0 + 8);
            b1.q[0] = *(const uint4*)(bp1); b1.q[1] = *(const uint4*)(bp1 + 8);
            b2.q[0] = *(const uint4*)(bp2); b2.q[1] = *(const uint4*)(bp2 + 8);
            b3.q[0] = *(const uint4*)(bp3); b3.q[1] = *(const uint4*)(bp3 + 8);

            acc[0][0] = __builtin_amdgcn_wmma_f32_16x16x32_bf16(false, a0.v, false, b0.v, (short)0, acc[0][0], false, false);
            acc[0][1] = __builtin_amdgcn_wmma_f32_16x16x32_bf16(false, a0.v, false, b1.v, (short)0, acc[0][1], false, false);
            acc[0][2] = __builtin_amdgcn_wmma_f32_16x16x32_bf16(false, a0.v, false, b2.v, (short)0, acc[0][2], false, false);
            acc[0][3] = __builtin_amdgcn_wmma_f32_16x16x32_bf16(false, a0.v, false, b3.v, (short)0, acc[0][3], false, false);
            acc[1][0] = __builtin_amdgcn_wmma_f32_16x16x32_bf16(false, a1.v, false, b0.v, (short)0, acc[1][0], false, false);
            acc[1][1] = __builtin_amdgcn_wmma_f32_16x16x32_bf16(false, a1.v, false, b1.v, (short)0, acc[1][1], false, false);
            acc[1][2] = __builtin_amdgcn_wmma_f32_16x16x32_bf16(false, a1.v, false, b2.v, (short)0, acc[1][2], false, false);
            acc[1][3] = __builtin_amdgcn_wmma_f32_16x16x32_bf16(false, a1.v, false, b3.v, (short)0, acc[1][3], false, false);
        }
        __syncthreads();   // protect buf from next iteration's async overwrite
    }

    #pragma unroll
    for (int r = 0; r < 2; ++r) {
        #pragma unroll
        for (int c = 0; c < 4; ++c) {
            float* yp = Y + (size_t)(rowBlk + wr * 32 + r * 16 + hi * 8) * N
                          + (colBlk + wc * 64 + c * 16 + laneLo);
            #pragma unroll
            for (int i = 0; i < 8; ++i) yp[(size_t)i * N] = acc[r][c][i];
        }
    }
}

// ---------------------------------------------------------------------------
// Block-wide sum reduction (256 threads, 8 wave32s)
// ---------------------------------------------------------------------------
__device__ __forceinline__ float block_reduce_sum(float v, float* smem) {
    #pragma unroll
    for (int o = 16; o > 0; o >>= 1) v += __shfl_xor(v, o, 32);
    const int wave = threadIdx.x >> 5;
    if ((threadIdx.x & 31) == 0) smem[wave] = v;
    __syncthreads();
    if (threadIdx.x == 0) {
        float t = 0.f;
        #pragma unroll
        for (int i = 0; i < 8; ++i) t += smem[i];
        smem[0] = t;
    }
    __syncthreads();
    float r = smem[0];
    __syncthreads();
    return r;
}

__device__ __forceinline__ float sigmoidf_(float t) {
    return 1.0f / (1.0f + __expf(-t));
}

// ---------------------------------------------------------------------------
// r-gate: r = sigmoid(LN(h@Wr; g0,b0) + LN(x@Ur; g1,b1));  hrb = bf16(h * r)
// ---------------------------------------------------------------------------
__global__ __launch_bounds__(256)
void rgate_kernel(const float* __restrict__ Y0, const float* __restrict__ Y1,
                  const float* __restrict__ h,
                  const float* __restrict__ gamma, const float* __restrict__ beta,
                  unsigned short* __restrict__ hrb) {
    __shared__ float smem[8];
    const int tid = threadIdx.x;
    const size_t base = (size_t)blockIdx.x * HH;

    float v0[4], v1[4];
    float s0 = 0.f, q0 = 0.f, s1 = 0.f, q1 = 0.f;
    #pragma unroll
    for (int j = 0; j < 4; ++j) {
        int col = tid + j * 256;
        v0[j] = Y0[base + col]; s0 += v0[j]; q0 += v0[j] * v0[j];
        v1[j] = Y1[base + col]; s1 += v1[j]; q1 += v1[j] * v1[j];
    }
    s0 = block_reduce_sum(s0, smem);
    q0 = block_reduce_sum(q0, smem);
    s1 = block_reduce_sum(s1, smem);
    q1 = block_reduce_sum(q1, smem);

    const float inv = 1.0f / (float)HH;
    float mu0 = s0 * inv, rs0 = rsqrtf(q0 * inv - mu0 * mu0 + 1e-5f);
    float mu1 = s1 * inv, rs1 = rsqrtf(q1 * inv - mu1 * mu1 + 1e-5f);

    #pragma unroll
    for (int j = 0; j < 4; ++j) {
        int col = tid + j * 256;
        float ln0 = (v0[j] - mu0) * rs0 * gamma[0 * HH + col] + beta[0 * HH + col];
        float ln1 = (v1[j] - mu1) * rs1 * gamma[1 * HH + col] + beta[1 * HH + col];
        float r = sigmoidf_(ln0 + ln1);
        hrb[base + col] = f32_to_bf16(h[base + col] * r);
    }
}

// ---------------------------------------------------------------------------
// final: u = sigmoid(LN(h@Wu; g2,b2) + LN(x@Uu; g3,b3))
//        c = tanh   (LN(hr@Wc; g4,b4) + LN(x@Uc; g5,b5))
//        out = (1-u)*h + u*c
// ---------------------------------------------------------------------------
__global__ __launch_bounds__(256)
void final_kernel(const float* __restrict__ Yuh, const float* __restrict__ Yux,
                  const float* __restrict__ Ych, const float* __restrict__ Ycx,
                  const float* __restrict__ h,
                  const float* __restrict__ gamma, const float* __restrict__ beta,
                  float* __restrict__ out) {
    __shared__ float smem[8];
    const int tid = threadIdx.x;
    const size_t base = (size_t)blockIdx.x * HH;

    float v2[4], v3[4], v4[4], v5[4];
    float s2 = 0.f, q2 = 0.f, s3 = 0.f, q3 = 0.f;
    float s4 = 0.f, q4 = 0.f, s5 = 0.f, q5 = 0.f;
    #pragma unroll
    for (int j = 0; j < 4; ++j) {
        int col = tid + j * 256;
        v2[j] = Yuh[base + col]; s2 += v2[j]; q2 += v2[j] * v2[j];
        v3[j] = Yux[base + col]; s3 += v3[j]; q3 += v3[j] * v3[j];
        v4[j] = Ych[base + col]; s4 += v4[j]; q4 += v4[j] * v4[j];
        v5[j] = Ycx[base + col]; s5 += v5[j]; q5 += v5[j] * v5[j];
    }
    s2 = block_reduce_sum(s2, smem); q2 = block_reduce_sum(q2, smem);
    s3 = block_reduce_sum(s3, smem); q3 = block_reduce_sum(q3, smem);
    s4 = block_reduce_sum(s4, smem); q4 = block_reduce_sum(q4, smem);
    s5 = block_reduce_sum(s5, smem); q5 = block_reduce_sum(q5, smem);

    const float inv = 1.0f / (float)HH;
    float mu2 = s2 * inv, rs2 = rsqrtf(q2 * inv - mu2 * mu2 + 1e-5f);
    float mu3 = s3 * inv, rs3 = rsqrtf(q3 * inv - mu3 * mu3 + 1e-5f);
    float mu4 = s4 * inv, rs4 = rsqrtf(q4 * inv - mu4 * mu4 + 1e-5f);
    float mu5 = s5 * inv, rs5 = rsqrtf(q5 * inv - mu5 * mu5 + 1e-5f);

    #pragma unroll
    for (int j = 0; j < 4; ++j) {
        int col = tid + j * 256;
        float ln2 = (v2[j] - mu2) * rs2 * gamma[2 * HH + col] + beta[2 * HH + col];
        float ln3 = (v3[j] - mu3) * rs3 * gamma[3 * HH + col] + beta[3 * HH + col];
        float ln4 = (v4[j] - mu4) * rs4 * gamma[4 * HH + col] + beta[4 * HH + col];
        float ln5 = (v5[j] - mu5) * rs5 * gamma[5 * HH + col] + beta[5 * HH + col];
        float u = sigmoidf_(ln2 + ln3);
        float c = tanhf(ln4 + ln5);
        float hv = h[base + col];
        out[base + col] = (1.0f - u) * hv + u * c;
    }
}

// ---------------------------------------------------------------------------
// launch
// ---------------------------------------------------------------------------
extern "C" void kernel_launch(void* const* d_in, const int* in_sizes, int n_in,
                              void* d_out, int out_size, void* d_ws, size_t ws_size,
                              hipStream_t stream) {
    const float* x   = (const float*)d_in[0];
    const float* h   = (const float*)d_in[1];
    const float* W[6] = {
        (const float*)d_in[2],   // W_r
        (const float*)d_in[3],   // U_r
        (const float*)d_in[4],   // W_u
        (const float*)d_in[5],   // U_u
        (const float*)d_in[6],   // W_c
        (const float*)d_in[7],   // U_c
    };
    const float* gamma = (const float*)d_in[8];
    const float* beta  = (const float*)d_in[9];
    float* out = (float*)d_out;

    const size_t BH = (size_t)BB * HH;   // = B*IN as well
    const size_t HK = (size_t)HH * KK;

    // workspace layout
    unsigned short* xb  = (unsigned short*)d_ws;      // B*K  bf16
    unsigned short* hb  = xb  + BH;                   // B*H  bf16
    unsigned short* hrb = hb  + BH;                   // B*H  bf16
    unsigned short* Wb  = hrb + BH;                   // 6*H*K bf16
    float* Ybase = (float*)(Wb + 6 * HK);             // 5 * B*H f32
    float* Y0 = Ybase + 0 * BH;   // h  @ W_r^T   (later reused for hr @ W_c^T)
    float* Y1 = Ybase + 1 * BH;   // x  @ U_r^T
    float* Y2 = Ybase + 2 * BH;   // h  @ W_u^T
    float* Y3 = Ybase + 3 * BH;   // x  @ U_u^T
    float* Y4 = Ybase + 4 * BH;   // x  @ U_c^T

    // 1) fp32 -> bf16 conversions
    {
        int n4 = (int)(BH / 4);
        cvt_f32_bf16<<<(n4 + 255) / 256, 256, 0, stream>>>(x, xb, n4);
        cvt_f32_bf16<<<(n4 + 255) / 256, 256, 0, stream>>>(h, hb, n4);
        int w4 = (int)(HK / 4);
        for (int i = 0; i < 6; ++i)
            cvt_f32_bf16<<<(w4 + 255) / 256, 256, 0, stream>>>(W[i], Wb + (size_t)i * HK, w4);
    }

    // 2) five independent GEMMs
    dim3 gg(HH / 128, BB / 128);
    gemm_bf16_wmma<<<gg, 256, 0, stream>>>(hb, Wb + 0 * HK, Y0, KK, HH);
    gemm_bf16_wmma<<<gg, 256, 0, stream>>>(xb, Wb + 1 * HK, Y1, KK, HH);
    gemm_bf16_wmma<<<gg, 256, 0, stream>>>(hb, Wb + 2 * HK, Y2, KK, HH);
    gemm_bf16_wmma<<<gg, 256, 0, stream>>>(xb, Wb + 3 * HK, Y3, KK, HH);
    gemm_bf16_wmma<<<gg, 256, 0, stream>>>(xb, Wb + 5 * HK, Y4, KK, HH);

    // 3) r gate + hr (bf16)
    rgate_kernel<<<BB, 256, 0, stream>>>(Y0, Y1, h, gamma, beta, hrb);

    // 4) (h*r) @ W_c^T  -> reuse slot 0
    gemm_bf16_wmma<<<gg, 256, 0, stream>>>(hrb, Wb + 4 * HK, Y0, KK, HH);

    // 5) u, c, output
    final_kernel<<<BB, 256, 0, stream>>>(Y2, Y3, Y0, Y4, h, gamma, beta, out);
}
